// VectorQuantizer_86715389706533
// MI455X (gfx1250) — compile-verified, matrix operand order checked by reading, not run on hardware
//
#include <hip/hip_runtime.h>
#include <hip/hip_bf16.h>

typedef __attribute__((ext_vector_type(16))) _Float16 v16h;
typedef __attribute__((ext_vector_type(8)))  _Float16 v8h;
typedef __attribute__((ext_vector_type(8)))  float    v8f;

#define DIM    256
#define NTOK   65536
#define CODES  1024
#define LDSA_STRIDE 264   // 256 + 8 halves pad -> 528B row stride, bank step 4

// d_out layout (all float, concatenated in reference return order)
#define OUT_OFF  0
#define IDX_OFF  (DIM * NTOK)          // 16777216
#define LOSS_OFF (IDX_OFF + NTOK)      // 16842752
#define DIST_OFF (LOSS_OFF + 1)        // 16842753

// ws layout (bytes):
//   [0,      4096)   e2: 1024 f32
//   [4096, 528384)   emb f16: 1024 x 256
//   [528384,544768)  per-block loss partials: 4096 f32

__global__ __launch_bounds__(256)
void vq_prep(const float* __restrict__ emb, float* __restrict__ e2,
             _Float16* __restrict__ embh) {
    __shared__ float red[256];
    const int c = blockIdx.x;
    const int t = threadIdx.x;
    const float v = emb[(size_t)c * DIM + t];
    embh[(size_t)c * DIM + t] = (_Float16)v;
    red[t] = v * v;
    __syncthreads();
    #pragma unroll
    for (int s = 128; s > 0; s >>= 1) {
        if (t < s) red[t] += red[t + s];
        __syncthreads();
    }
    if (t == 0) e2[c] = red[0];
}

__global__ __launch_bounds__(256)
void vq_main(const float* __restrict__ x,        // inputs (D, N)
             const float* __restrict__ emb,      // (C, D) fp32
             const _Float16* __restrict__ embh,  // (C, D) f16
             const float* __restrict__ e2,       // (C)
             float* __restrict__ out,            // d_out base
             float* __restrict__ partials) {
    __shared__ _Float16 Atile[16 * LDSA_STRIDE];
    __shared__ float x2s[16];
    __shared__ float redv[16][128];
    __shared__ int   redc[16][128];
    __shared__ int   rowidx[16];
    __shared__ float lred[256];

    const int t    = threadIdx.x;
    const int tok0 = blockIdx.x * 16;

    if (t < 16) x2s[t] = 0.0f;
    __syncthreads();

    // ---- Stage A tile (16 tokens x 256 dims) fp32 -> f16 LDS; accumulate x^2 in fp32
    {
        const int m = t & 15;   // token within strip
        const int g = t >> 4;   // k-group: covers k in [g*16, g*16+16)
        float ssq = 0.0f;
        #pragma unroll
        for (int j = 0; j < 16; ++j) {
            const int k = g * 16 + j;
            const float v = x[(size_t)k * NTOK + tok0 + m];  // 64B coalesced per 16 lanes
            ssq += v * v;
            Atile[m * LDSA_STRIDE + k] = (_Float16)v;
        }
        atomicAdd(&x2s[m], ssq);   // ds_add_f32
    }
    __syncthreads();

    const int lane = t & 31;
    const int wv   = t >> 5;       // wave 0..7 -> 128-column chunk
    const int m16  = lane & 15;
    const int hi   = lane >> 4;    // half-wave selector per WMMA layout

    float x2loc[8];
    #pragma unroll
    for (int i = 0; i < 8; ++i) x2loc[i] = x2s[hi * 8 + i];

    float bestv[8];
    int   bestc[8];
    #pragma unroll
    for (int i = 0; i < 8; ++i) { bestv[i] = -3.4e38f; bestc[i] = 0; }

    float* __restrict__ dist = out + DIST_OFF;

    #pragma unroll 1
    for (int tt = 0; tt < 8; ++tt) {
        const int col = wv * 128 + tt * 16 + m16;
        v8f acc = {0.f, 0.f, 0.f, 0.f, 0.f, 0.f, 0.f, 0.f};
        // 16-bit operand layout: lo lanes hold K{0..7,16..23}, hi lanes K{8..15,24..31}
        const _Float16* ap = Atile + m16 * LDSA_STRIDE + hi * 8;         // A row = token m16
        const _Float16* bp = embh + (size_t)col * DIM + hi * 8;          // B col = code col
        #pragma unroll
        for (int kk = 0; kk < 8; ++kk) {
            v8h a0 = *(const v8h*)(ap);        // ds_load_b128
            v8h a1 = *(const v8h*)(ap + 16);
            v8h b0 = *(const v8h*)(bp);        // global_load_b128 (L2-resident)
            v8h b1 = *(const v8h*)(bp + 16);
            v16h a = __builtin_shufflevector(a0, a1, 0,1,2,3,4,5,6,7,8,9,10,11,12,13,14,15);
            v16h b = __builtin_shufflevector(b0, b1, 0,1,2,3,4,5,6,7,8,9,10,11,12,13,14,15);
            acc = __builtin_amdgcn_wmma_f32_16x16x32_f16(
                false, a, false, b, (short)0, acc, false, false);
            ap += 32;
            bp += 32;
        }
        const float e2c = e2[col];
        #pragma unroll
        for (int i = 0; i < 8; ++i) {
            const int row = hi * 8 + i;        // C/D layout: VGPR i <-> rows i / 8+i
            float d2 = x2loc[i] + e2c - 2.0f * acc[i];
            d2 = fmaxf(d2, 0.0f);
            const float dv = -sqrtf(d2);
            // streaming write-once output: non-temporal so it doesn't evict
            // the L2-resident codebook/inputs (192MB L2 vs 256MB of distances)
            __builtin_nontemporal_store(dv, &dist[(size_t)(tok0 + row) * CODES + col]);
            if (dv > bestv[i] || (dv == bestv[i] && col < bestc[i])) {
                bestv[i] = dv; bestc[i] = col;
            }
        }
    }

    // ---- Cross-wave argmax per token row
    #pragma unroll
    for (int i = 0; i < 8; ++i) {
        redv[hi * 8 + i][wv * 16 + m16] = bestv[i];
        redc[hi * 8 + i][wv * 16 + m16] = bestc[i];
    }
    __syncthreads();
    if (t < 16) {
        float bv = -3.4e38f; int bc = 0;
        for (int j = 0; j < 128; ++j) {
            const float v = redv[t][j];
            const int   c = redc[t][j];
            if (v > bv || (v == bv && c < bc)) { bv = v; bc = c; }
        }
        rowidx[t] = bc;
        out[IDX_OFF + tok0 + t] = (float)bc;   // indices as float in concat output
    }
    __syncthreads();

    // ---- Gather quantized rows (fp32 codebook) + commitment-loss partial
    {
        const int m = t & 15;
        const int g = t >> 4;
        const int idx = rowidx[m];
        float ls = 0.0f;
        #pragma unroll
        for (int j = 0; j < 16; ++j) {
            const int d = g * 16 + j;
            const float q  = emb[(size_t)idx * DIM + d];
            const float xv = x[(size_t)d * NTOK + tok0 + m];
            // quantized_st == quantized numerically; streaming store, non-temporal
            __builtin_nontemporal_store(q, &out[OUT_OFF + (size_t)d * NTOK + tok0 + m]);
            const float df = q - xv;
            ls += df * df;
        }
        lred[t] = ls;
    }
    __syncthreads();
    #pragma unroll
    for (int s = 128; s > 0; s >>= 1) {
        if (t < s) lred[t] += lred[t + s];
        __syncthreads();
    }
    if (t == 0)
        partials[blockIdx.x] = lred[0] * (0.25f / ((float)DIM * (float)NTOK));
}

__global__ __launch_bounds__(256)
void vq_loss(const float* __restrict__ partials, float* __restrict__ out) {
    __shared__ float red[256];
    const int t = threadIdx.x;
    float s = 0.0f;
    #pragma unroll
    for (int j = 0; j < 16; ++j) s += partials[t + 256 * j];
    red[t] = s;
    __syncthreads();
    #pragma unroll
    for (int k = 128; k > 0; k >>= 1) {
        if (t < k) red[t] += red[t + k];
        __syncthreads();
    }
    if (t == 0) out[LOSS_OFF] = red[0];
}

extern "C" void kernel_launch(void* const* d_in, const int* in_sizes, int n_in,
                              void* d_out, int out_size, void* d_ws, size_t ws_size,
                              hipStream_t stream) {
    const float* x   = (const float*)d_in[0];   // inputs (1, 256, 65536) fp32
    const float* emb = (const float*)d_in[1];   // embedding (1, 1024, 256) fp32
    float* out = (float*)d_out;

    char* ws = (char*)d_ws;
    float*    e2       = (float*)ws;                                    // 4 KB
    _Float16* embh     = (_Float16*)(ws + 4096);                        // 512 KB
    float*    partials = (float*)(ws + 4096 + (size_t)CODES * DIM * 2); // 16 KB

    hipLaunchKernelGGL(vq_prep, dim3(CODES), dim3(256), 0, stream, emb, e2, embh);
    hipLaunchKernelGGL(vq_main, dim3(NTOK / 16), dim3(256), 0, stream,
                       x, emb, embh, e2, out, partials);
    hipLaunchKernelGGL(vq_loss, dim3(1), dim3(256), 0, stream, partials, out);
}
